// APPNPNet_28991029248694
// MI455X (gfx1250) — matrix operand order repaired and checked by reading, not compiled, not run on hardware
//
#include <hip/hip_runtime.h>
#include <hip/hip_bf16.h>
#include <math.h>

#define NN 100000
#define EE 3200000
#define FF 512
#define DD 256
#define CC 64
#define KITER 10
#define ALPHA 0.1f

typedef __bf16 bf16_t;
typedef bf16_t v8bf  __attribute__((ext_vector_type(8)));
typedef bf16_t v16bf __attribute__((ext_vector_type(16)));
typedef float  v8f   __attribute__((ext_vector_type(8)));

__device__ __forceinline__ float sigmoidf_(float v) {
    return 1.0f / (1.0f + __expf(-v));
}

// Build bf16 A-fragment from fp32 row with per-feature scale.
// Lane layout used here (contiguous k, B loaded the same way): lane&15 = M; elems 0..7 = K=koff..koff+7,
// elems 8..15 = K=koff+8..koff+15, koff = (lane>>4)*16.
__device__ __forceinline__ v16bf makeA_f32(const float* __restrict__ arow,
                                           const float* __restrict__ xs,
                                           int k, int koff) {
    v16bf a;
#pragma unroll
    for (int i = 0; i < 8; ++i) {
        a[i]     = (bf16_t)(arow[k + koff + i]      * xs[k + koff + i]);
        a[i + 8] = (bf16_t)(arow[k + 8 + koff + i] * xs[k + 8 + koff + i]);
    }
    return a;
}

__device__ __forceinline__ v16bf makeA_bf16(const bf16_t* __restrict__ arow,
                                            int k, int koff) {
    v8bf lo = *(const v8bf*)(arow + k + koff);
    v8bf hi = *(const v8bf*)(arow + k + 8 + koff);
    return __builtin_shufflevector(lo, hi, 0,1,2,3,4,5,6,7,8,9,10,11,12,13,14,15);
}

// ---------------- prep kernels ----------------

__global__ void k_xws(const float* __restrict__ xw, float* __restrict__ xs) {
    int i = blockIdx.x * blockDim.x + threadIdx.x;
    if (i >= FF) return;
    float w = xw[i];
    xs[i] = (fabsf(w) > 0.0f) ? sigmoidf_(w) : 0.0f;
}

__global__ void k_w1t(const float* __restrict__ W1, bf16_t* __restrict__ w1t) {
    int id = blockIdx.x * blockDim.x + threadIdx.x;      // FF*DD
    if (id >= FF * DD) return;
    int n = id / FF, k = id - n * FF;
    w1t[(size_t)n * FF + k] = (bf16_t)W1[(size_t)k * DD + n];
}

__global__ void k_w2t(const float* __restrict__ W2, bf16_t* __restrict__ w2t) {
    int id = blockIdx.x * blockDim.x + threadIdx.x;      // DD*CC
    if (id >= DD * CC) return;
    int n = id / DD, k = id - n * DD;
    w2t[(size_t)n * DD + k] = (bf16_t)W2[(size_t)k * CC + n];
}

__global__ void k_deg_init(float* __restrict__ deg) {
    int i = blockIdx.x * blockDim.x + threadIdx.x;
    if (i < NN) deg[i] = 1.0f;                           // self-loop weight
}

__global__ void k_deg(const long long* __restrict__ ei, const float* __restrict__ ew,
                      float* __restrict__ deg) {
    int e = blockIdx.x * blockDim.x + threadIdx.x;
    if (e >= EE) return;
    float t = ew[e];
    float w = (fabsf(t) > 0.0f) ? sigmoidf_(t) : 0.0f;
    long long c = ei[(size_t)EE + e];
    unsafeAtomicAdd(&deg[c], w);
}

__global__ void k_dinv(float* __restrict__ deg) {
    int i = blockIdx.x * blockDim.x + threadIdx.x;
    if (i >= NN) return;
    float d = deg[i];
    deg[i] = (d > 0.0f) ? rsqrtf(fmaxf(d, 1e-12f)) : 0.0f;
}

__global__ void k_norm(const long long* __restrict__ ei, const float* __restrict__ ew,
                       const float* __restrict__ dinv, float* __restrict__ nrm) {
    int e = blockIdx.x * blockDim.x + threadIdx.x;
    if (e >= EE) return;
    float t = ew[e];
    float w = (fabsf(t) > 0.0f) ? sigmoidf_(t) : 0.0f;
    long long r = ei[e], c = ei[(size_t)EE + e];
    nrm[e] = dinv[r] * w * dinv[c];
}

// ---------------- GEMM1: h = relu((x .* xs) @ W1 + b1) ----------------
// wave -> 16 rows x 128 cols (8 tiles); gridDim.y = 2 column groups.
// Register double-buffering: next k-step A/B loaded before current WMMAs.

__global__ __launch_bounds__(128) void k_gemm1(
    const float* __restrict__ x, const float* __restrict__ xs,
    const bf16_t* __restrict__ w1t, const float* __restrict__ b1,
    bf16_t* __restrict__ h)
{
    const int lane = threadIdx.x & 31;
    const int tile = blockIdx.x * 4 + (threadIdx.x >> 5);
    if (tile >= NN / 16) return;
    const int cg   = blockIdx.y;                 // 0 or 1: 128-col group
    const int mlo  = lane & 15;
    const int half = lane >> 4;
    const int koff = half * 16;
    const float*  arow  = x + (size_t)(tile * 16 + mlo) * FF;
    const bf16_t* bbase = w1t + (size_t)(cg * 128 + mlo) * FF + half * 16;

    v8f acc[8] = {};

    v16bf aCur = makeA_f32(arow, xs, 0, koff);
    v16bf bCur[8];
#pragma unroll
    for (int nt = 0; nt < 8; ++nt)
        bCur[nt] = *(const v16bf*)(bbase + (size_t)nt * 16 * FF);

#pragma unroll
    for (int k = 0; k < FF; k += 32) {
        const int kn = (k + 32 < FF) ? (k + 32) : 0;   // wrap: harmless prefetch
        v16bf aNext = makeA_f32(arow, xs, kn, koff);
        v16bf bNext[8];
#pragma unroll
        for (int nt = 0; nt < 8; ++nt)
            bNext[nt] = *(const v16bf*)(bbase + (size_t)nt * 16 * FF + kn);
#pragma unroll
        for (int nt = 0; nt < 8; ++nt)
            acc[nt] = __builtin_amdgcn_wmma_f32_16x16x32_bf16(
                false, aCur, false, bCur[nt], (short)0, acc[nt], false, false);
        aCur = aNext;
#pragma unroll
        for (int nt = 0; nt < 8; ++nt) bCur[nt] = bNext[nt];
    }

    const int mhi = half * 8;
#pragma unroll
    for (int nt = 0; nt < 8; ++nt) {
        const int n = cg * 128 + nt * 16 + mlo;
        const float bias = b1[n];
#pragma unroll
        for (int r = 0; r < 8; ++r) {
            float v = acc[nt][r] + bias;
            v = v > 0.0f ? v : 0.0f;
            h[(size_t)(tile * 16 + mhi + r) * DD + n] = (bf16_t)v;
        }
    }
}

// ---------------- GEMM2: z0 = h @ W2 + b2 ----------------

__global__ __launch_bounds__(128) void k_gemm2(
    const bf16_t* __restrict__ h, const bf16_t* __restrict__ w2t,
    const float* __restrict__ b2, float* __restrict__ z0)
{
    const int lane = threadIdx.x & 31;
    const int tile = blockIdx.x * 4 + (threadIdx.x >> 5);
    if (tile >= NN / 16) return;
    const int mlo  = lane & 15;
    const int half = lane >> 4;
    const int koff = half * 16;
    const bf16_t* arow  = h + (size_t)(tile * 16 + mlo) * DD;
    const bf16_t* bbase = w2t + (size_t)mlo * DD + half * 16;

    v8f acc[4] = {};

    v16bf aCur = makeA_bf16(arow, 0, koff);
    v16bf bCur[4];
#pragma unroll
    for (int nt = 0; nt < 4; ++nt)
        bCur[nt] = *(const v16bf*)(bbase + (size_t)nt * 16 * DD);

#pragma unroll
    for (int k = 0; k < DD; k += 32) {
        const int kn = (k + 32 < DD) ? (k + 32) : 0;
        v16bf aNext = makeA_bf16(arow, kn, koff);
        v16bf bNext[4];
#pragma unroll
        for (int nt = 0; nt < 4; ++nt)
            bNext[nt] = *(const v16bf*)(bbase + (size_t)nt * 16 * DD + kn);
#pragma unroll
        for (int nt = 0; nt < 4; ++nt)
            acc[nt] = __builtin_amdgcn_wmma_f32_16x16x32_bf16(
                false, aCur, false, bCur[nt], (short)0, acc[nt], false, false);
        aCur = aNext;
#pragma unroll
        for (int nt = 0; nt < 4; ++nt) bCur[nt] = bNext[nt];
    }

    const int mhi = half * 8;
#pragma unroll
    for (int nt = 0; nt < 4; ++nt) {
        const int n = nt * 16 + mlo;
        const float bias = b2[n];
#pragma unroll
        for (int r = 0; r < 8; ++r)
            z0[(size_t)(tile * 16 + mhi + r) * CC + n] = acc[nt][r] + bias;
    }
}

// ---------------- propagation ----------------

__global__ void k_selfinit(const float* __restrict__ z, const float* __restrict__ dinv,
                           float* __restrict__ agg) {
    int t = blockIdx.x * blockDim.x + threadIdx.x;       // NN*CC
    if (t >= NN * CC) return;
    int i = t >> 6;
    float di = dinv[i];
    agg[t] = z[t] * di * di;
}

__global__ void k_scatter(const long long* __restrict__ ei, const float* __restrict__ nrm,
                          const float* __restrict__ z, float* __restrict__ agg) {
    unsigned t = blockIdx.x * blockDim.x + threadIdx.x;  // EE*16
    unsigned e = t >> 4;
    if (e >= EE) return;
    int q = (t & 15) * 4;
    float w = nrm[e];
    long long r = ei[e], c = ei[(size_t)EE + e];
    float4 zv = *(const float4*)(z + (size_t)r * CC + q);
    float* dst = agg + (size_t)c * CC + q;
    unsafeAtomicAdd(dst + 0, zv.x * w);
    unsafeAtomicAdd(dst + 1, zv.y * w);
    unsafeAtomicAdd(dst + 2, zv.z * w);
    unsafeAtomicAdd(dst + 3, zv.w * w);
}

__global__ void k_combine(const float* __restrict__ agg, const float* __restrict__ z0,
                          float* __restrict__ zn) {
    int t = blockIdx.x * blockDim.x + threadIdx.x;       // NN*CC
    if (t >= NN * CC) return;
    zn[t] = (1.0f - ALPHA) * agg[t] + ALPHA * z0[t];
}

// ---------------- log_softmax over C=64, one wave per node ----------------

__global__ void k_logsoftmax(const float* __restrict__ z, float* __restrict__ out) {
    int lane = threadIdx.x & 31;
    int node = blockIdx.x * 8 + (threadIdx.x >> 5);
    if (node >= NN) return;
    const float* zr = z + (size_t)node * CC;
    float2 v = *(const float2*)(zr + lane * 2);
    float m = fmaxf(v.x, v.y);
#pragma unroll
    for (int off = 16; off; off >>= 1) m = fmaxf(m, __shfl_xor(m, off, 32));
    float s = __expf(v.x - m) + __expf(v.y - m);
#pragma unroll
    for (int off = 16; off; off >>= 1) s += __shfl_xor(s, off, 32);
    float lse = m + __logf(s);
    float2 o; o.x = v.x - lse; o.y = v.y - lse;
    *(float2*)(out + (size_t)node * CC + lane * 2) = o;
}

// ---------------- host launcher ----------------

extern "C" void kernel_launch(void* const* d_in, const int* in_sizes, int n_in,
                              void* d_out, int out_size, void* d_ws, size_t ws_size,
                              hipStream_t stream) {
    const float*     x   = (const float*)d_in[0];
    const long long* ei  = (const long long*)d_in[1];
    const float*     ew  = (const float*)d_in[2];
    const float*     xw  = (const float*)d_in[3];
    const float*     W1  = (const float*)d_in[4];
    const float*     b1  = (const float*)d_in[5];
    const float*     W2  = (const float*)d_in[6];
    const float*     b2  = (const float*)d_in[7];
    float* out = (float*)d_out;

    char* p = (char*)d_ws;
    auto carve = [&](size_t bytes) -> void* {
        void* r = (void*)p;
        p += (bytes + 255) & ~(size_t)255;
        return r;
    };
    float*  xs   = (float*) carve((size_t)FF * 4);
    float*  dinv = (float*) carve((size_t)NN * 4);
    float*  nrm  = (float*) carve((size_t)EE * 4);
    bf16_t* w1t  = (bf16_t*)carve((size_t)FF * DD * 2);
    bf16_t* w2t  = (bf16_t*)carve((size_t)DD * CC * 2);
    bf16_t* h    = (bf16_t*)carve((size_t)NN * DD * 2);
    float*  z0   = (float*) carve((size_t)NN * CC * 4);
    float*  za   = (float*) carve((size_t)NN * CC * 4);
    float*  zb   = (float*) carve((size_t)NN * CC * 4);
    float*  agg  = (float*) carve((size_t)NN * CC * 4);

    k_xws     <<<(FF + 255) / 256, 256, 0, stream>>>(xw, xs);
    k_w1t     <<<(FF * DD + 255) / 256, 256, 0, stream>>>(W1, w1t);
    k_w2t     <<<(DD * CC + 255) / 256, 256, 0, stream>>>(W2, w2t);
    k_deg_init<<<(NN + 255) / 256, 256, 0, stream>>>(dinv);
    k_deg     <<<EE / 256, 256, 0, stream>>>(ei, ew, dinv);
    k_dinv    <<<(NN + 255) / 256, 256, 0, stream>>>(dinv);
    k_norm    <<<EE / 256, 256, 0, stream>>>(ei, ew, dinv, nrm);

    const int ntile = NN / 16;                       // 6250
    dim3 g1((ntile + 3) / 4, 2);
    k_gemm1<<<g1, 128, 0, stream>>>(x, xs, w1t, b1, h);
    k_gemm2<<<(ntile + 3) / 4, 128, 0, stream>>>(h, w2t, b2, z0);

    const float* zc = z0;
    float* bufs[2] = { za, zb };
    for (int it = 0; it < KITER; ++it) {
        float* zn = bufs[it & 1];
        k_selfinit<<<(NN * CC) / 256, 256, 0, stream>>>(zc, dinv, agg);
        k_scatter <<<(EE * 16) / 256, 256, 0, stream>>>(ei, nrm, zc, agg);
        k_combine <<<(NN * CC) / 256, 256, 0, stream>>>(agg, z0, zn);
        zc = zn;
    }
    k_logsoftmax<<<NN / 8, 256, 0, stream>>>(zc, out);
}
